// HeterogeneousGraphSAGE_51891794870357
// MI455X (gfx1250) — matrix operand order repaired and checked by reading, not compiled, not run on hardware
//
#include <hip/hip_runtime.h>
#include <hip/hip_bf16.h>

typedef __attribute__((ext_vector_type(16))) __bf16 v16bf;
typedef __attribute__((ext_vector_type(8)))  __bf16 v8bf;
typedef __attribute__((ext_vector_type(2)))  __bf16 v2bf;
typedef __attribute__((ext_vector_type(8)))  float  v8f;

#define N_USER 100000
#define N_GAME 20000
#define NEDGE  800000
#define NPAIR  400000
#define H 128
#define FSTRIDE 24   // bf16 units per lane slot: 16 data + 8 pad (48B, 16B-aligned)

// ---------------------------------------------------------------------------
// Sparse side: degree count, inverse, scatter-add (segment_sum numerator)
// ---------------------------------------------------------------------------
__global__ void count_deg_kernel(const int* __restrict__ dst, float* __restrict__ cnt, int E) {
    int i = blockIdx.x * blockDim.x + threadIdx.x;
    if (i < E) atomicAdd(&cnt[dst[i]], 1.0f);
}

__global__ void inv_cnt_kernel(const float* __restrict__ cnt, float* __restrict__ inv, int n) {
    int i = blockIdx.x * blockDim.x + threadIdx.x;
    if (i < n) inv[i] = 1.0f / fmaxf(cnt[i], 1.0f);
}

// One thread moves 4 consecutive floats of one edge's source row.
__global__ void scatter_add_kernel(const float* __restrict__ xsrc,
                                   const int* __restrict__ src,
                                   const int* __restrict__ dst,
                                   float* __restrict__ acc,
                                   int E, int dim, int perShift) {
    const int per = 1 << perShift;                 // dim/4
    long long idx = (long long)blockIdx.x * blockDim.x + threadIdx.x;
    long long total = (long long)E * per;
    if (idx >= total) return;
    int e = (int)(idx >> perShift);
    int c = ((int)idx & (per - 1)) << 2;
    int s = src[e], d = dst[e];
    const float4 v = *(const float4*)(xsrc + (size_t)s * dim + c);
    float* p = acc + (size_t)d * dim + c;
    atomicAdd(p + 0, v.x); atomicAdd(p + 1, v.y);
    atomicAdd(p + 2, v.z); atomicAdd(p + 3, v.w);
}

// ---------------------------------------------------------------------------
// Fragment helpers: LDS holds tiles pre-swizzled into WMMA lane layout so a
// fragment load is two contiguous ds_load_b128 per lane (no scalar packing).
// ---------------------------------------------------------------------------
__device__ __forceinline__ v16bf load_frag(const __bf16* base) {
    v8bf lo = *(const v8bf*)(base);
    v8bf hi = *(const v8bf*)(base + 8);
    return __builtin_shufflevector(lo, hi, 0, 1, 2, 3, 4, 5, 6, 7,
                                   8, 9, 10, 11, 12, 13, 14, 15);
}

// A (16x32 bf16 per wave-strip): slot/elem for element (row m, k)
__device__ __forceinline__ int a_slot(int m, int k) {
    return (m >> 4) * 32 + ((k >> 4) << 4) + (m & 15);
}
__device__ __forceinline__ int a_elem(int k) {
    return k & 15;
}
// B (32x16 bf16 per col tile): slot/elem for element (k, col n)
__device__ __forceinline__ int b_slot(int n, int k) {
    return (n >> 4) * 32 + ((k >> 4) << 4) + (n & 15);
}

// ---------------------------------------------------------------------------
// Fused SAGE update: out[N,128] = (acc/deg)@Wl + b + Xdst@Wr  (optional ReLU)
// bf16 WMMA 16x16x32. Block = 256 threads (8 waves) -> 128x128 output tile.
// ---------------------------------------------------------------------------
template<bool RELU>
__global__ __launch_bounds__(256, 2)
void sage_gemm_kernel(const float* __restrict__ accA, const float* __restrict__ invc,
                      const float* __restrict__ xdst,
                      const float* __restrict__ wl, const float* __restrict__ bias,
                      const float* __restrict__ wr,
                      float* __restrict__ out, int N, int Kl, int Kr) {
    __shared__ __attribute__((aligned(16))) __bf16 Af[8 * 32 * FSTRIDE];
    __shared__ __attribute__((aligned(16))) __bf16 Bf[8 * 32 * FSTRIDE];

    const int tid  = threadIdx.x;
    const int lane = tid & 31;
    const int wave = tid >> 5;
    const int half = lane >> 4;
    const int l16  = lane & 15;
    const int row0 = blockIdx.x * 128;

    v8f acc[8];
#pragma unroll
    for (int t = 0; t < 8; ++t)
#pragma unroll
        for (int j = 0; j < 8; ++j) acc[t][j] = 0.0f;

    for (int pass = 0; pass < 2; ++pass) {
        const float* Ap = pass ? xdst : accA;
        const float* Wp = pass ? wr   : wl;
        const int    K  = pass ? Kr   : Kl;
        for (int k0 = 0; k0 < K; k0 += 32) {
            // A tile 128x32: fp32->bf16 pairs along k, mean-scale on pass 0.
            for (int i = tid; i < 128 * 16; i += 256) {
                int r = i >> 4, c = (i & 15) << 1;
                int gr = row0 + r;
                float v0 = 0.0f, v1 = 0.0f;
                if (gr < N) {
                    const float* p = Ap + (size_t)gr * K + k0 + c;
                    v0 = p[0]; v1 = p[1];
                    if (!pass) { float s = invc[gr]; v0 *= s; v1 *= s; }
                }
                v2bf pk = {(__bf16)v0, (__bf16)v1};
                *(v2bf*)&Af[a_slot(r, c) * FSTRIDE + a_elem(c)] = pk;
            }
            // B tile 32x128: pairs along k, lanes sweep columns (coalesced).
            for (int i = tid; i < 16 * 128; i += 256) {
                int kp = i >> 7, c = i & 127;
                int k  = kp << 1;
                float v0 = Wp[(size_t)(k0 + k) * H + c];
                float v1 = Wp[(size_t)(k0 + k + 1) * H + c];
                v2bf pk = {(__bf16)v0, (__bf16)v1};
                *(v2bf*)&Bf[b_slot(c, k) * FSTRIDE + (k & 15)] = pk;
            }
            __syncthreads();

            v16bf a = load_frag(&Af[(wave * 32 + lane) * FSTRIDE]);
#pragma unroll
            for (int t = 0; t < 8; ++t) {
                v16bf b = load_frag(&Bf[(t * 32 + lane) * FSTRIDE]);
                acc[t] = __builtin_amdgcn_wmma_f32_16x16x32_bf16(
                    false, a, false, b, (short)0, acc[t], false, false);
            }
            __syncthreads();
        }
    }

    // Epilogue: C layout VGPR r -> rows r (lanes0-15) / r+8 (lanes16-31)
    const int mbase = row0 + wave * 16;
#pragma unroll
    for (int t = 0; t < 8; ++t) {
        const int col = t * 16 + l16;
        const float bc = bias[col];
#pragma unroll
        for (int r = 0; r < 8; ++r) {
            int row = mbase + r + half * 8;
            if (row < N) {
                float v = acc[t][r] + bc;
                if (RELU) v = fmaxf(v, 0.0f);
                out[(size_t)row * H + col] = v;
            }
        }
    }
}

// ---------------------------------------------------------------------------
// Fused decoder: z = [u3[row], g3[col]] (P x 256); h = relu(z@W1+b1);
// out = h@w2 + b2. Second matmul done in-register via 16-lane shfl reduce.
// ---------------------------------------------------------------------------
__global__ __launch_bounds__(256, 2)
void decoder_kernel(const float* __restrict__ u3, const float* __restrict__ g3,
                    const int* __restrict__ lab_row, const int* __restrict__ lab_col,
                    const float* __restrict__ w1, const float* __restrict__ b1,
                    const float* __restrict__ w2, const float* __restrict__ b2,
                    float* __restrict__ out, int P) {
    __shared__ __attribute__((aligned(16))) __bf16 Af[8 * 32 * FSTRIDE];
    __shared__ __attribute__((aligned(16))) __bf16 Bf[8 * 32 * FSTRIDE];
    __shared__ int rIdx[128];
    __shared__ int cIdx[128];

    const int tid  = threadIdx.x;
    const int lane = tid & 31;
    const int wave = tid >> 5;
    const int half = lane >> 4;
    const int l16  = lane & 15;
    const int row0 = blockIdx.x * 128;

    if (tid < 128) {
        int p = row0 + tid;
        int pc = p < P ? p : (P - 1);
        rIdx[tid] = lab_row[pc];
        cIdx[tid] = lab_col[pc];
    }
    __syncthreads();

    v8f acc[8];
#pragma unroll
    for (int t = 0; t < 8; ++t)
#pragma unroll
        for (int j = 0; j < 8; ++j) acc[t][j] = 0.0f;

    for (int k0 = 0; k0 < 256; k0 += 32) {
        // Gathered A tile (concat of u3[row] and g3[col] features), k-pairs.
        for (int i = tid; i < 128 * 16; i += 256) {
            int r = i >> 4, c = (i & 15) << 1;
            int k = k0 + c;
            const float* sp = (k < 128) ? (u3 + (size_t)rIdx[r] * H + k)
                                        : (g3 + (size_t)cIdx[r] * H + (k - 128));
            v2bf pk = {(__bf16)sp[0], (__bf16)sp[1]};
            *(v2bf*)&Af[a_slot(r, c) * FSTRIDE + a_elem(c)] = pk;
        }
        for (int i = tid; i < 16 * 128; i += 256) {
            int kp = i >> 7, c = i & 127;
            int k  = kp << 1;
            float v0 = w1[(size_t)(k0 + k) * H + c];
            float v1 = w1[(size_t)(k0 + k + 1) * H + c];
            v2bf pk = {(__bf16)v0, (__bf16)v1};
            *(v2bf*)&Bf[b_slot(c, k) * FSTRIDE + (k & 15)] = pk;
        }
        __syncthreads();

        v16bf a = load_frag(&Af[(wave * 32 + lane) * FSTRIDE]);
#pragma unroll
        for (int t = 0; t < 8; ++t) {
            v16bf b = load_frag(&Bf[(t * 32 + lane) * FSTRIDE]);
            acc[t] = __builtin_amdgcn_wmma_f32_16x16x32_bf16(
                false, a, false, b, (short)0, acc[t], false, false);
        }
        __syncthreads();
    }

    // Fused layer 2: per-row dot of relu(h) with w2[128], bias b1 folded in.
    float partial[8];
#pragma unroll
    for (int r = 0; r < 8; ++r) partial[r] = 0.0f;
#pragma unroll
    for (int t = 0; t < 8; ++t) {
        const int col = t * 16 + l16;
        const float b1c = b1[col];
        const float w2c = w2[col];
#pragma unroll
        for (int r = 0; r < 8; ++r) {
            float hv = fmaxf(acc[t][r] + b1c, 0.0f);
            partial[r] += hv * w2c;
        }
    }
    // Reduce across the 16-lane column group (offsets 1,2,4,8 stay in-half)
#pragma unroll
    for (int off = 8; off > 0; off >>= 1)
#pragma unroll
        for (int r = 0; r < 8; ++r)
            partial[r] += __shfl_xor(partial[r], off, 32);

    if (l16 == 0) {
        const float bb = b2[0];
#pragma unroll
        for (int r = 0; r < 8; ++r) {
            int p = row0 + wave * 16 + r + half * 8;
            if (p < P) out[p] = partial[r] + bb;
        }
    }
}

// ---------------------------------------------------------------------------
// Host orchestration
// ---------------------------------------------------------------------------
static inline size_t alignup(size_t x) { return (x + 255) & ~(size_t)255; }

extern "C" void kernel_launch(void* const* d_in, const int* in_sizes, int n_in,
                              void* d_out, int out_size, void* d_ws, size_t ws_size,
                              hipStream_t stream) {
    const float* x_user = (const float*)d_in[0];
    const float* x_game = (const float*)d_in[1];
    const float* w1ug_l = (const float*)d_in[2];
    const float* b1ug_l = (const float*)d_in[3];
    const float* w1ug_r = (const float*)d_in[4];
    const float* w1gu_l = (const float*)d_in[5];
    const float* b1gu_l = (const float*)d_in[6];
    const float* w1gu_r = (const float*)d_in[7];
    const float* w2ug_l = (const float*)d_in[8];
    const float* b2ug_l = (const float*)d_in[9];
    const float* w2ug_r = (const float*)d_in[10];
    const float* w2gu_l = (const float*)d_in[11];
    const float* b2gu_l = (const float*)d_in[12];
    const float* w2gu_r = (const float*)d_in[13];
    const float* w3ug_l = (const float*)d_in[14];
    const float* b3ug_l = (const float*)d_in[15];
    const float* w3ug_r = (const float*)d_in[16];
    const float* w3gu_l = (const float*)d_in[17];
    const float* b3gu_l = (const float*)d_in[18];
    const float* w3gu_r = (const float*)d_in[19];
    const float* dec_w1 = (const float*)d_in[20];
    const float* dec_b1 = (const float*)d_in[21];
    const float* dec_w2 = (const float*)d_in[22];
    const float* dec_b2 = (const float*)d_in[23];
    const int* src_ug = (const int*)d_in[24];
    const int* dst_ug = (const int*)d_in[25];
    const int* src_gu = (const int*)d_in[26];
    const int* dst_gu = (const int*)d_in[27];
    const int* lab_row = (const int*)d_in[28];
    const int* lab_col = (const int*)d_in[29];
    float* out = (float*)d_out;

    // Workspace carve-up
    char* ws = (char*)d_ws;
    size_t off = 0;
    float* cnt_u  = (float*)(ws + off); off = alignup(off + (size_t)N_USER * 4);
    float* cnt_g  = (float*)(ws + off); off = alignup(off + (size_t)N_GAME * 4);
    float* invc_u = (float*)(ws + off); off = alignup(off + (size_t)N_USER * 4);
    float* invc_g = (float*)(ws + off); off = alignup(off + (size_t)N_GAME * 4);
    float* acc_u  = (float*)(ws + off); off = alignup(off + (size_t)N_USER * H * 4);
    float* acc_g  = (float*)(ws + off); off = alignup(off + (size_t)N_GAME * H * 4);
    float* u_a    = (float*)(ws + off); off = alignup(off + (size_t)N_USER * H * 4);
    float* u_b    = (float*)(ws + off); off = alignup(off + (size_t)N_USER * H * 4);
    float* g_a    = (float*)(ws + off); off = alignup(off + (size_t)N_GAME * H * 4);
    float* g_b    = (float*)(ws + off); off = alignup(off + (size_t)N_GAME * H * 4);
    (void)ws_size; (void)n_in; (void)in_sizes; (void)out_size;

    const int T = 256;
    const int ebk = (NEDGE + T - 1) / T;

    // Degrees (graph-invariant within a call; recomputed for determinism)
    hipMemsetAsync(cnt_u, 0, (size_t)N_USER * 4, stream);
    hipMemsetAsync(cnt_g, 0, (size_t)N_GAME * 4, stream);
    count_deg_kernel<<<ebk, T, 0, stream>>>(dst_ug, cnt_g, NEDGE);
    count_deg_kernel<<<ebk, T, 0, stream>>>(dst_gu, cnt_u, NEDGE);
    inv_cnt_kernel<<<(N_USER + T - 1) / T, T, 0, stream>>>(cnt_u, invc_u, N_USER);
    inv_cnt_kernel<<<(N_GAME + T - 1) / T, T, 0, stream>>>(cnt_g, invc_g, N_GAME);

    const int gG = (N_GAME + 127) / 128;
    const int gU = (N_USER + 127) / 128;

    auto nblk = [](long long total, int t) { return (int)((total + t - 1) / t); };

    // ---- Layer 1 ----
    hipMemsetAsync(acc_g, 0, (size_t)N_GAME * 64 * 4, stream);
    scatter_add_kernel<<<nblk((long long)NEDGE * 16, T), T, 0, stream>>>(
        x_user, src_ug, dst_ug, acc_g, NEDGE, 64, 4);
    sage_gemm_kernel<true><<<gG, T, 0, stream>>>(
        acc_g, invc_g, x_game, w1ug_l, b1ug_l, w1ug_r, g_a, N_GAME, 64, 32);

    hipMemsetAsync(acc_u, 0, (size_t)N_USER * 32 * 4, stream);
    scatter_add_kernel<<<nblk((long long)NEDGE * 8, T), T, 0, stream>>>(
        x_game, src_gu, dst_gu, acc_u, NEDGE, 32, 3);
    sage_gemm_kernel<true><<<gU, T, 0, stream>>>(
        acc_u, invc_u, x_user, w1gu_l, b1gu_l, w1gu_r, u_a, N_USER, 32, 64);

    // ---- Layer 2 ----
    hipMemsetAsync(acc_g, 0, (size_t)N_GAME * H * 4, stream);
    scatter_add_kernel<<<nblk((long long)NEDGE * 32, T), T, 0, stream>>>(
        u_a, src_ug, dst_ug, acc_g, NEDGE, H, 5);
    sage_gemm_kernel<true><<<gG, T, 0, stream>>>(
        acc_g, invc_g, g_a, w2ug_l, b2ug_l, w2ug_r, g_b, N_GAME, H, H);

    hipMemsetAsync(acc_u, 0, (size_t)N_USER * H * 4, stream);
    scatter_add_kernel<<<nblk((long long)NEDGE * 32, T), T, 0, stream>>>(
        g_a, src_gu, dst_gu, acc_u, NEDGE, H, 5);
    sage_gemm_kernel<true><<<gU, T, 0, stream>>>(
        acc_u, invc_u, u_a, w2gu_l, b2gu_l, w2gu_r, u_b, N_USER, H, H);

    // ---- Layer 3 (no ReLU) ----
    hipMemsetAsync(acc_g, 0, (size_t)N_GAME * H * 4, stream);
    scatter_add_kernel<<<nblk((long long)NEDGE * 32, T), T, 0, stream>>>(
        u_b, src_ug, dst_ug, acc_g, NEDGE, H, 5);
    sage_gemm_kernel<false><<<gG, T, 0, stream>>>(
        acc_g, invc_g, g_b, w3ug_l, b3ug_l, w3ug_r, g_a, N_GAME, H, H);

    hipMemsetAsync(acc_u, 0, (size_t)N_USER * H * 4, stream);
    scatter_add_kernel<<<nblk((long long)NEDGE * 32, T), T, 0, stream>>>(
        g_b, src_gu, dst_gu, acc_u, NEDGE, H, 5);
    sage_gemm_kernel<false><<<gU, T, 0, stream>>>(
        acc_u, invc_u, u_b, w3gu_l, b3gu_l, w3gu_r, u_a, N_USER, H, H);

    // ---- Decoder ----
    decoder_kernel<<<(NPAIR + 127) / 128, T, 0, stream>>>(
        u_a, g_a, lab_row, lab_col, dec_w1, dec_b1, dec_w2, dec_b2, out, NPAIR);
}
